// orient_21689584844960
// MI455X (gfx1250) — matrix operand order repaired and checked by reading, not compiled, not run on hardware
//
#include <hip/hip_runtime.h>
#include <stdint.h>

typedef __attribute__((ext_vector_type(16))) _Float16 v16h;
typedef __attribute__((ext_vector_type(8)))  float    v8f;
typedef __attribute__((ext_vector_type(4)))  int      v4i;
typedef __attribute__((ext_vector_type(4)))  float    v4f;
typedef __attribute__((ext_vector_type(4)))  unsigned int u32x4;
typedef __attribute__((ext_vector_type(8)))  int      i32x8;
typedef __attribute__((ext_vector_type(4)))  int      i32x4;

#define NK    32              // oriented kernels (output channels)
#define KS    17              // kernel spatial size
#define KXP   32              // kx padded to 32: one WMMA K-chunk == one filter row
#define PADR  8
#define NB    8               // batch
#define S     512             // image H == W
#define TR    4               // output rows per workgroup
#define TC    32              // output cols per workgroup
#define IRO   (TR + KS - 1)   // 20 input rows in LDS tile
#define ICO   64              // input cols in LDS tile
#define WELEMS (NK * KS * KXP)          // 17408 f16
#define WBYTES (WELEMS * 2)             // 34816 B == 4352 x 8B

// ---------------------------------------------------------------------------
// Kernel 1: build the 32 oriented DoG filters as f16 [32][17][32] (kx>=17 = 0)
// ---------------------------------------------------------------------------
__global__ void build_filters(_Float16* __restrict__ w) {
  int idx = blockIdx.x * 256 + threadIdx.x;
  if (idx >= WELEMS) return;
  int k   = idx / (KS * KXP);
  int rem = idx % (KS * KXP);
  int i   = rem / KXP;       // ky  (reference x = r[i])
  int j   = rem % KXP;       // kx  (reference y = r[j])
  float val = 0.0f;
  if (j < KS) {
    float theta = (float)k * (3.14159265358979323846f / (float)NK);
    float sn, cs;
    __sincosf(theta, &sn, &cs);
    float x = (float)(i - PADR);
    float y = (float)(j - PADR);
    float xt =  x * cs + y * sn;
    float yt = -x * sn + y * cs;
    // SIGMA_H=1, SIGMA_L=2, SIGMA_Y=2; 1/(1/SH - 1/SL) = 2
    float g1 = __expf(-0.5f * (xt * xt         + yt * yt * 0.25f));
    float g2 = __expf(-0.5f * (xt * xt * 0.25f + yt * yt * 0.25f)) * 0.5f;
    val = (g1 - g2) * 2.0f;
  }
  w[idx] = (_Float16)val;
}

// ---------------------------------------------------------------------------
// Kernel 2: implicit-GEMM conv via v_wmma_f32_16x16x32_f16 + fused relu/argmax
// ---------------------------------------------------------------------------
struct A32 { v4i lo, hi; };   // 32 bytes == one v16h A/B fragment

__device__ __forceinline__ void async_b128(uint32_t lds_off, uint32_t goff,
                                           uint64_t sbase) {
  // GVS mode: mem_addr = SADDR + VADDR(u32); LDS dst byte offset in first vgpr.
  asm volatile("global_load_async_to_lds_b128 %0, %1, %2"
               :: "v"(lds_off), "v"(goff), "s"(sbase) : "memory");
}

__global__ __launch_bounds__(256) void dog_conv(const float* __restrict__ img,
                                                const _Float16* __restrict__ wglob,
                                                float* __restrict__ out) {
  __shared__ __attribute__((aligned(16))) _Float16 wlds[WELEMS];
  __shared__ __attribute__((aligned(16))) float    tile[IRO][ICO];

  const int tid  = threadIdx.x;
  const int col0 = blockIdx.x * TC;
  const int row0 = blockIdx.y * TR;
  const int b    = blockIdx.z;

  // ---- stage filter bank (34816 B, contiguous) via Tensor Data Mover ------
  // One 1D D# descriptor: 4352 elements of 8 bytes. Issued by wave 0 only
  // (TDM ignores EXEC; per-wave issue would duplicate the DMA 8x).
  if (tid < 32) {
    const uint64_t gaddr = (uint64_t)(uintptr_t)wglob;
    const uint32_t laddr = (uint32_t)(uintptr_t)&wlds[0];
    u32x4 g0;
    g0.x = 1u;                                   // count=1, no gather/restore
    g0.y = laddr;                                // lds_addr [63:32]
    g0.z = (uint32_t)gaddr;                      // global_addr[31:0]
    g0.w = (uint32_t)(gaddr >> 32) | 0x80000000u;// global_addr[56:32] | type=2
    i32x8 g1;
    g1[0] = 3 << 16;          // workgroup_mask=0, data_size=3 (8B), no flags
    g1[1] = (int)((WBYTES / 8) << 16);  // tensor_dim0[15:0] (=4352) in [31:16]
    g1[2] = 1 << 16;          // tensor_dim0 hi=0 ; tensor_dim1=1 (lo 16 bits)
    g1[3] = (int)((WBYTES / 8) << 16);  // tensor_dim1 hi=0 ; tile_dim0=4352
    g1[4] = 0;                // tile_dim1=0 (unused), tile_dim2=0
    g1[5] = (int)(WBYTES / 8);// tensor_dim0_stride[31:0]
    g1[6] = 0;                // stride0 hi, stride1 lo
    g1[7] = 0;                // stride1 hi
    i32x4 z4 = {0, 0, 0, 0};
    i32x8 z8 = {0, 0, 0, 0, 0, 0, 0, 0};
    __builtin_amdgcn_tensor_load_to_lds(g0, g1, z4, z4, z8, 0);
    __builtin_amdgcn_s_wait_tensorcnt(0);
  }

  // ---- stage image tile (f32) with zero-padded borders: async copy --------
  // pad(8) and S(512) are multiples of 4 -> every 16B chunk is fully in/out.
  {
    const uint64_t ibase = (uint64_t)(uintptr_t)img;
    const uint32_t tl0   = (uint32_t)(uintptr_t)&tile[0][0];
    for (int cid = tid; cid < IRO * (ICO / 4); cid += 256) {
      int r   = cid >> 4;          // tile row
      int j   = cid & 15;          // 16B chunk within row
      int iy  = row0 - PADR + r;
      int ix0 = col0 - PADR + j * 4;
      if (iy >= 0 && iy < S && ix0 >= 0 && ix0 + 3 < S) {
        uint32_t goff = (uint32_t)((((b * S) + iy) * S + ix0) * 4);
        async_b128(tl0 + (uint32_t)cid * 16u, goff, ibase);
      } else {
        v4f z = {0.f, 0.f, 0.f, 0.f};
        *(v4f*)&tile[r][j * 4] = z;   // disjoint from async-written chunks
      }
    }
  }

  asm volatile("s_wait_asynccnt 0x0" ::: "memory");
  __syncthreads();

  // ---- per-wave 16x16 output tile: M=channels(2 tiles of 16), N=16 pixels -
  const int lane = tid & 31;
  const int w    = tid >> 5;       // 8 waves
  const int oyl  = w >> 1;         // 0..3 : output row within block tile
  const int oxh  = (w & 1) << 4;   // 0/16 : output col half within block tile
  const int nc   = lane & 15;      // N index (pixel)
  const int half = lane >> 4;      // lane half selects K sub-block

  v8f acc0 = {0.f, 0.f, 0.f, 0.f, 0.f, 0.f, 0.f, 0.f};
  v8f acc1 = {0.f, 0.f, 0.f, 0.f, 0.f, 0.f, 0.f, 0.f};

  // A-frag base (f16 elements): channel*KS*KXP + half*8 ; per-ky offset is a
  // compile-time immediate after full unroll.
  const _Float16* wpA = &wlds[(nc       * KS) * KXP + half * 8];
  const _Float16* wpB = &wlds[((nc + 16) * KS) * KXP + half * 8];
  const int cb = oxh + nc + half * 8;    // B-frag start column in tile row

#pragma unroll
  for (int ky = 0; ky < KS; ++ky) {
    const int r = oyl + ky;          // input row in tile
    // A fragments (16-bit A 16x32 layout, ISA 7.12.2):
    //  lane half -> K+8 ; vgprs 0-3 at kx0, vgprs 4-7 at kx0+16
    A32 t0, t1;
    t0.lo = *(const v4i*)(wpA + ky * KXP);
    t0.hi = *(const v4i*)(wpA + ky * KXP + 16);
    t1.lo = *(const v4i*)(wpB + ky * KXP);
    t1.hi = *(const v4i*)(wpB + ky * KXP + 16);
    v16h a0 = __builtin_bit_cast(v16h, t0);
    v16h a1 = __builtin_bit_cast(v16h, t1);

    // B fragment (16-bit B 32x16): lane half -> K+8, elements 0-7 at K,
    // elements 8-15 at K+16 (matches A layout); f32->f16 cvt in VALU
    // co-executes with the XDL WMMAs.
    v16h bf;
#pragma unroll
    for (int e = 0; e < 16; ++e) bf[e] = (_Float16)tile[r][cb + e + (e >= 8 ? 8 : 0)];

    acc0 = __builtin_amdgcn_wmma_f32_16x16x32_f16(false, a0, false, bf,
                                                  (short)0, acc0, false, false);
    acc1 = __builtin_amdgcn_wmma_f32_16x16x32_f16(false, a1, false, bf,
                                                  (short)0, acc1, false, false);
  }

  // ---- fused ReLU + argmax/max over 32 channels ---------------------------
  // Lane holds channels half*8+r (acc0) and 16+half*8+r (acc1) for pixel nc.
  float best; int bidx;
  {
    const int c0 = half * 8;
    best = fmaxf(acc0[0], 0.0f);
    bidx = c0;
#pragma unroll
    for (int rr = 1; rr < 8; ++rr) {
      float v = fmaxf(acc0[rr], 0.0f);
      if (v > best) { best = v; bidx = c0 + rr; }
    }
#pragma unroll
    for (int rr = 0; rr < 8; ++rr) {
      float v = fmaxf(acc1[rr], 0.0f);
      if (v > best) { best = v; bidx = 16 + c0 + rr; }
    }
  }
  // combine the two lane-halves (channels interleave), lowest index wins ties
  float pval = __shfl_xor(best, 16, 32);
  int   pidx = __shfl_xor(bidx, 16, 32);
  if (pval > best || (pval == best && pidx < bidx)) { best = pval; bidx = pidx; }

  if (half == 0) {
    const int oy = row0 + oyl;
    const int ox = col0 + oxh + nc;
    const int o  = (b * S + oy) * S + ox;
    out[o]              = (float)bidx;  // argmax
    out[NB * S * S + o] = best;         // confidence (relu'd max)
  }
}

// ---------------------------------------------------------------------------
extern "C" void kernel_launch(void* const* d_in, const int* in_sizes, int n_in,
                              void* d_out, int out_size, void* d_ws, size_t ws_size,
                              hipStream_t stream) {
  const float* image = (const float*)d_in[0];
  _Float16*    wbank = (_Float16*)d_ws;      // needs WBYTES (34816 B) of scratch
  float*       out   = (float*)d_out;

  build_filters<<<(WELEMS + 255) / 256, 256, 0, stream>>>(wbank);
  dog_conv<<<dim3(S / TC, S / TR, NB), 256, 0, stream>>>(image, wbank, out);
}
